// DepMamba_77206332113511
// MI455X (gfx1250) — matrix operand order repaired and checked by reading, not compiled, not run on hardware
//
#include <hip/hip_runtime.h>

// ---------------------------------------------------------------------------
// DepMamba forward for gfx1250 (MI455X): bf16 WMMA GEMMs (+ TDM-staged
// weights for the dense projections) + selective scan.
// ---------------------------------------------------------------------------

typedef __bf16        v16bf __attribute__((ext_vector_type(16)));
typedef float         v8f   __attribute__((ext_vector_type(8)));
typedef unsigned int  u32x4 __attribute__((ext_vector_type(4)));
typedef int           i32x8 __attribute__((ext_vector_type(8)));
typedef int           i32x4 __attribute__((ext_vector_type(4)));

#if defined(__has_builtin)
#  if __has_builtin(__builtin_amdgcn_tensor_load_to_lds) && \
      __has_builtin(__builtin_amdgcn_s_wait_tensorcnt)
#    define HAVE_TDM 1
#  endif
#endif
#ifndef HAVE_TDM
#  define HAVE_TDM 0
#endif

static __device__ __forceinline__ __bf16 f2bf(float f) {
  return static_cast<__bf16>(f);   // native f32->bf16 (RNE on gfx1250)
}

// ---------------------------------------------------------------------------
// Dense GEMM (K%32==0, no shift, no accumulate):
//   Out[M,N] = A[M,K] * W[N,K]^T, f32 in/out, bf16 WMMA compute.
// Block = 4 waves = 128 rows x 16 cols. All 4 waves share one weight tile
// (16 rows x 32 K, f32) which is DMA'd into LDS by the Tensor Data Mover,
// double-buffered (TENSORcnt), with native TDM padding (4 dwords per 32)
// giving a 36-dword LDS pitch (bank-conflict-free ds_load_b128 reads).
// ---------------------------------------------------------------------------
#define WTILE_DW 36                     // padded LDS row pitch in dwords
#define WTILE_BYTES (16 * WTILE_DW * 4) // 2304 bytes per buffer

static __device__ __forceinline__ void tdm_issue_w_tile(
    const float* gptr, long ldw, unsigned lds_byte)
{
#if HAVE_TDM
  const unsigned long long ga = (unsigned long long)(size_t)gptr;
  u32x4 g0;
  g0.x = 1u;                                        // count=1 (valid), user D#
  g0.y = lds_byte;                                  // LDS dest (bytes)
  g0.z = (unsigned)(ga & 0xFFFFFFFFu);              // global_addr[31:0]
  g0.w = (unsigned)((ga >> 32) & 0x01FFFFFFu)       // global_addr[56:32]
       | (2u << 30);                                // type = 2 ("image")
  i32x8 g1;
  g1[0] = (int)((2u << 16)                          // data_size = 4B
              | (1u << 20)                          // pad_enable
              | (4u << 22)                          // pad_interval: 32 dwords
              | (3u << 25));                        // pad_amount: 4 dwords
  g1[1] = (int)(32u << 16);                         // tensor_dim0 = 32
  g1[2] = (int)(16u << 16);                         // tensor_dim1 = 16
  g1[3] = (int)(32u << 16);                         // tile_dim0 = 32
  g1[4] = 16;                                       // tile_dim1 = 16
  g1[5] = (int)(unsigned)ldw;                       // dim0 stride (elements)
  g1[6] = 0;
  g1[7] = 0;
  const i32x4 gz4 = {0, 0, 0, 0};
  const i32x8 gz8 = {0, 0, 0, 0, 0, 0, 0, 0};
  __builtin_amdgcn_tensor_load_to_lds(g0, g1, gz4, gz4, gz8, 0);
#else
  (void)gptr; (void)ldw; (void)lds_byte;
#endif
}

__global__ __launch_bounds__(128) void gemm_bf16_wmma_tdm(
    const float* __restrict__ A, int lda,
    const float* __restrict__ W, long ldw,
    float* __restrict__ Out, int ldo,
    int M, int N, int K)
{
  __shared__ float ldsW[2][16 * WTILE_DW];          // only LDS object: offset 0

  const int lane   = threadIdx.x & 31;
  const int wave   = threadIdx.x >> 5;
  const int tilesN = N >> 4;
  const int tn     = blockIdx.x % tilesN;
  const int tmb    = blockIdx.x / tilesN;           // 128-row block index

  const int mrow = lane & 15;
  const int lh   = lane >> 4;

  const int r0 = tmb * 128 + wave * 32 + mrow;
  const float* __restrict__ A0 = A + (long)r0 * lda;
  const float* __restrict__ A1 = A0 + (long)16 * lda;
  const float* __restrict__ Wt = W + (long)(tn * 16) * ldw;

  v8f acc0, acc1;
#pragma unroll
  for (int i = 0; i < 8; ++i) { acc0[i] = 0.0f; acc1[i] = 0.0f; }

#if HAVE_TDM
  if (threadIdx.x < 32) tdm_issue_w_tile(Wt, ldw, (unsigned)(size_t)&ldsW[0][0]);
#endif

  int it = 0;
  for (int kb = 0; kb < K; kb += 32, ++it) {
    const int cur = it & 1;
#if HAVE_TDM
    if (threadIdx.x < 32) {
      if (kb + 32 < K) {
        tdm_issue_w_tile(Wt + (kb + 32), ldw,
                         (unsigned)(size_t)&ldsW[cur ^ 1][0]);
        __builtin_amdgcn_s_wait_tensorcnt(1);   // current tile complete
      } else {
        __builtin_amdgcn_s_wait_tensorcnt(0);   // drain last tile
      }
    }
#else
    // Cooperative fallback: 128 threads copy the 16x32 tile (padded pitch).
    for (int idx = threadIdx.x; idx < 16 * 8; idx += 128) {
      const int row = idx >> 3, seg = idx & 7;
      const float4 v = *(const float4*)(Wt + (long)row * ldw + kb + seg * 4);
      *(float4*)&ldsW[cur][row * WTILE_DW + seg * 4] = v;
    }
#endif
    __syncthreads();

    // A fragments: direct vectorized loads (rows are always in-bounds here).
    if (kb + 64 < K) {
      __builtin_prefetch(A0 + kb + 64, 0, 1);
      __builtin_prefetch(A1 + kb + 64, 0, 1);
    }
    const float4* q0 = (const float4*)(A0 + kb + (lh << 3));
    const float4* q1 = (const float4*)(A1 + kb + (lh << 3));
    float ta[16], tb[16];
    *(float4*)(ta + 0)  = q0[0]; *(float4*)(ta + 4)  = q0[1];
    *(float4*)(ta + 8)  = q0[4]; *(float4*)(ta + 12) = q0[5];
    *(float4*)(tb + 0)  = q1[0]; *(float4*)(tb + 4)  = q1[1];
    *(float4*)(tb + 8)  = q1[4]; *(float4*)(tb + 12) = q1[5];

    // W fragment from LDS (lane = column, K at lh*8+0..7 and 16+lh*8+0..7).
    const float* wp = &ldsW[cur][mrow * WTILE_DW + (lh << 3)];
    float tw[16];
    *(float4*)(tw + 0)  = *(const float4*)(wp + 0);
    *(float4*)(tw + 4)  = *(const float4*)(wp + 4);
    *(float4*)(tw + 8)  = *(const float4*)(wp + 16);
    *(float4*)(tw + 12) = *(const float4*)(wp + 20);

    v16bf a0, a1, w;
#pragma unroll
    for (int e = 0; e < 16; ++e) {
      a0[e] = f2bf(ta[e]);
      a1[e] = f2bf(tb[e]);
      w[e]  = f2bf(tw[e]);
    }

    acc0 = __builtin_amdgcn_wmma_f32_16x16x32_bf16(
               false, a0, false, w, (short)0, acc0, false, false);
    acc1 = __builtin_amdgcn_wmma_f32_16x16x32_bf16(
               false, a1, false, w, (short)0, acc1, false, false);

    __syncthreads();   // all waves done with ldsW[cur] before it is reloaded
  }

  const int rowb = tmb * 128 + wave * 32;
#pragma unroll
  for (int i = 0; i < 8; ++i) {
    Out[(long)(rowb + i + lh * 8) * ldo + tn * 16 + mrow]      = acc0[i];
    Out[(long)(rowb + 16 + i + lh * 8) * ldo + tn * 16 + mrow] = acc1[i];
  }
}

// ---------------------------------------------------------------------------
// General GEMM (conv taps / small K): value-masked (multiply) loads, no
// exec-divergent control flow. FASTK: K%32==0 vector A loads.
// ---------------------------------------------------------------------------
template <bool FASTK>
__global__ __launch_bounds__(128) void gemm_bf16_wmma_gen(
    const float* __restrict__ A, int lda,
    const float* __restrict__ W, long ldw, int wstride,
    float* __restrict__ Out, int ldo,
    int M, int N, int K, int L, int shift, int accumulate)
{
  const int lane   = threadIdx.x & 31;
  const int wave   = threadIdx.x >> 5;
  const int tilesN = N >> 4;
  const int t      = blockIdx.x * 4 + wave;
  const int tm     = t / tilesN;          // 32-row block index
  const int tn     = t - tm * tilesN;

  const int mrow = lane & 15;
  const int lh   = lane >> 4;

  const int r0 = tm * 32 + mrow;
  const int r1 = r0 + 16;
  const int b0 = r0 / L, l0 = r0 - b0 * L;
  const int b1 = r1 / L, l1 = r1 - b1 * L;
  const int s0 = l0 + shift, s1 = l1 + shift;
  const float m0 = ((s0 >= 0) && (s0 < L)) ? 1.0f : 0.0f;
  const float m1 = ((s1 >= 0) && (s1 < L)) ? 1.0f : 0.0f;
  const int c0 = s0 < 0 ? 0 : (s0 >= L ? L - 1 : s0);
  const int c1 = s1 < 0 ? 0 : (s1 >= L ? L - 1 : s1);
  const float* __restrict__ A0 = A + (long)(b0 * L + c0) * lda;
  const float* __restrict__ A1 = A + (long)(b1 * L + c1) * lda;

  const int wn = tn * 16 + mrow;
  const float* __restrict__ Wr = W + (long)wn * ldw;

  v8f acc0, acc1;
  if (accumulate) {
#pragma unroll
    for (int i = 0; i < 8; ++i) {
      acc0[i] = Out[(long)(tm * 32 + i + lh * 8) * ldo + tn * 16 + mrow];
      acc1[i] = Out[(long)(tm * 32 + 16 + i + lh * 8) * ldo + tn * 16 + mrow];
    }
  } else {
#pragma unroll
    for (int i = 0; i < 8; ++i) { acc0[i] = 0.0f; acc1[i] = 0.0f; }
  }

  for (int kb = 0; kb < K; kb += 32) {
    v16bf a0, a1, w;

    if (FASTK) {
      const float4* q0 = (const float4*)(A0 + kb + (lh << 3));
      const float4* q1 = (const float4*)(A1 + kb + (lh << 3));
      float ta[16], tb[16];
      *(float4*)(ta + 0)  = q0[0]; *(float4*)(ta + 4)  = q0[1];
      *(float4*)(ta + 8)  = q0[4]; *(float4*)(ta + 12) = q0[5];
      *(float4*)(tb + 0)  = q1[0]; *(float4*)(tb + 4)  = q1[1];
      *(float4*)(tb + 8)  = q1[4]; *(float4*)(tb + 12) = q1[5];
#pragma unroll
      for (int e = 0; e < 16; ++e) {
        a0[e] = f2bf(ta[e] * m0);
        a1[e] = f2bf(tb[e] * m1);
      }
    } else {
#pragma unroll
      for (int e = 0; e < 16; ++e) {
        const int v = e >> 1, p = e & 1;
        const int ka  = kb + ((v >> 2) << 4) + ((v & 3) << 1) + p + (lh << 3);
        const int kac = ka < K ? ka : 0;
        const float km = ka < K ? 1.0f : 0.0f;
        a0[e] = f2bf(A0[kac] * (m0 * km));
        a1[e] = f2bf(A1[kac] * (m1 * km));
      }
    }

#pragma unroll
    for (int e = 0; e < 16; ++e) {
      const int v  = e >> 1, p = e & 1;
      const int kw = kb + ((v >> 2) << 4) + ((v & 3) << 1) + p + (lh << 3);
      const int kwc = FASTK ? kw : (kw < K ? kw : 0);
      const float km = FASTK ? 1.0f : (kw < K ? 1.0f : 0.0f);
      w[e] = f2bf(Wr[(long)kwc * wstride] * km);
    }

    acc0 = __builtin_amdgcn_wmma_f32_16x16x32_bf16(
               false, a0, false, w, (short)0, acc0, false, false);
    acc1 = __builtin_amdgcn_wmma_f32_16x16x32_bf16(
               false, a1, false, w, (short)0, acc1, false, false);
  }

#pragma unroll
  for (int i = 0; i < 8; ++i) {
    Out[(long)(tm * 32 + i + lh * 8) * ldo + tn * 16 + mrow]      = acc0[i];
    Out[(long)(tm * 32 + 16 + i + lh * 8) * ldo + tn * 16 + mrow] = acc1[i];
  }
}

// ---------------------------------------------------------------------------
// Front end: xe = x[:,:,:32]@w_eeg^T, xg = x[:,:,32]@w_gsr^T, then per-modality
// conv1d(k=3,same) + eval-BN + ReLU + 1x1 skip, writing e,g (M x 256).
// ---------------------------------------------------------------------------
__global__ __launch_bounds__(256) void frontend_kernel(
    const float* __restrict__ x,
    const float* __restrict__ w_eeg, const float* __restrict__ w_gsr,
    const float* __restrict__ e_conv_w, const float* __restrict__ g_conv_w,
    const float* __restrict__ e_bn_g, const float* __restrict__ e_bn_b,
    const float* __restrict__ g_bn_g, const float* __restrict__ g_bn_b,
    const float* __restrict__ e_skip_w, const float* __restrict__ g_skip_w,
    float* __restrict__ e_out, float* __restrict__ g_out, int L)
{
  const int r = blockIdx.x;
  const int b = r / L;
  const int l = r - b * L;
  __shared__ float se[3][16];
  __shared__ float sg[3][16];
  const int t = threadIdx.x;
  if (t < 48) {
    const int pos = t >> 4;          // 0,1,2 -> l-1, l, l+1
    const int ch  = t & 15;
    const int lsrc = l + pos - 1;
    float ve = 0.0f, vg = 0.0f;
    if (lsrc >= 0 && lsrc < L) {
      const float* xr = x + (long)(b * L + lsrc) * 33;
#pragma unroll
      for (int i = 0; i < 32; ++i) ve += xr[i] * w_eeg[ch * 32 + i];
      vg = xr[32] * w_gsr[ch];
    }
    se[pos][ch] = ve;
    sg[pos][ch] = vg;
  }
  __syncthreads();

  const int co = t;
  float ce = 0.0f, cg = 0.0f, ske = 0.0f, skg = 0.0f;
#pragma unroll
  for (int ch = 0; ch < 16; ++ch) {
#pragma unroll
    for (int s = 0; s < 3; ++s) {
      ce += se[s][ch] * e_conv_w[(co * 16 + ch) * 3 + s];
      cg += sg[s][ch] * g_conv_w[(co * 16 + ch) * 3 + s];
    }
    ske += se[1][ch] * e_skip_w[co * 16 + ch];
    skg += sg[1][ch] * g_skip_w[co * 16 + ch];
  }
  const float sc = rsqrtf(1.0f + 1e-5f);
  e_out[(long)r * 256 + co] = fmaxf(ce * e_bn_g[co] * sc + e_bn_b[co], 0.0f) + ske;
  g_out[(long)r * 256 + co] = fmaxf(cg * g_bn_g[co] * sc + g_bn_b[co], 0.0f) + skg;
}

// Causal depthwise conv (k=4, left pad 3) + bias + SiLU.
__global__ __launch_bounds__(256) void dwconv_silu_kernel(
    const float* __restrict__ xi, const float* __restrict__ cw,
    const float* __restrict__ cb, float* __restrict__ xc,
    int L, int di, long total)
{
  const long i = (long)blockIdx.x * 256 + threadIdx.x;
  if (i >= total) return;
  const int  d = (int)(i % di);
  const long r = i / di;
  const int  l = (int)(r % (long)L);
  float acc = cb[d];
#pragma unroll
  for (int j = 0; j < 4; ++j) {
    const int lsrc = l - 3 + j;
    const long rs = (lsrc >= 0) ? (r - 3 + j) : r;
    const float m = (lsrc >= 0) ? 1.0f : 0.0f;
    acc += xi[rs * di + d] * m * cw[d * 4 + j];
  }
  xc[i] = acc / (1.0f + __expf(-acc));
}

// dt = softplus(dt_raw + bias[d]) in place.
__global__ __launch_bounds__(256) void bias_softplus_kernel(
    float* __restrict__ dt, const float* __restrict__ bias, int di, long total)
{
  const long i = (long)blockIdx.x * 256 + threadIdx.x;
  if (i >= total) return;
  const float v = dt[i] + bias[(int)(i % di)];
  dt[i] = (v > 20.0f) ? v : log1pf(__expf(v));
}

// ---------------------------------------------------------------------------
// Selective scan. grid = (di/256, B); block = 256 threads, one channel each.
// ---------------------------------------------------------------------------
__global__ __launch_bounds__(256) void scan_kernel(
    const float* __restrict__ dt, const float* __restrict__ xc,
    const float* __restrict__ dbl, int nd, int dtr,
    const float* __restrict__ A_log, const float* __restrict__ Dp,
    const float* __restrict__ z, float* __restrict__ y,
    int L, int di)
{
  const int d = blockIdx.x * 256 + threadIdx.x;
  const int b = blockIdx.y;
  float h[16];
  float Ar[16];
#pragma unroll
  for (int s = 0; s < 16; ++s) { h[s] = 0.0f; Ar[s] = -__expf(A_log[d * 16 + s]); }
  const float dpar = Dp[d];

  __shared__ float sB[16];
  __shared__ float sC[16];

  for (int l = 0; l < L; ++l) {
    const long r = (long)(b * L + l);
    if (threadIdx.x < 32) {
      const float v = dbl[r * nd + dtr + threadIdx.x];
      if (threadIdx.x < 16) sB[threadIdx.x] = v;
      else                  sC[threadIdx.x - 16] = v;
    }
    __syncthreads();
    const float dtv = dt[r * di + d];
    const float xv  = xc[r * di + d];
    const float dx  = dtv * xv;
    float acc = 0.0f;
#pragma unroll
    for (int s = 0; s < 16; ++s) {
      h[s] = h[s] * __expf(dtv * Ar[s]) + dx * sB[s];
      acc += h[s] * sC[s];
    }
    const float zz = z[r * di + d];
    y[r * di + d] = (acc + xv * dpar) * (zz / (1.0f + __expf(-zz)));
    __syncthreads();
  }
}

// LayerNorm of yin row + add into residual stream hio (in place).
__global__ __launch_bounds__(256) void ln_residual_kernel(
    const float* __restrict__ yin, const float* __restrict__ g,
    const float* __restrict__ bb, float* __restrict__ hio, int D)
{
  const int r = blockIdx.x;
  const float* row = yin + (long)r * D;
  float* hrow = hio + (long)r * D;
  __shared__ float s1[256];
  __shared__ float s2[256];
  const int t = threadIdx.x;
  float s = 0.0f, sq = 0.0f;
  for (int c = t; c < D; c += 256) { const float v = row[c]; s += v; sq += v * v; }
  s1[t] = s; s2[t] = sq;
  __syncthreads();
  for (int o = 128; o > 0; o >>= 1) {
    if (t < o) { s1[t] += s1[t + o]; s2[t] += s2[t + o]; }
    __syncthreads();
  }
  const float mean = s1[0] / (float)D;
  const float var  = s2[0] / (float)D - mean * mean;
  const float inv  = rsqrtf(var + 1e-6f);
  for (int c = t; c < D; c += 256)
    hrow[c] += (row[c] - mean) * inv * g[c] + bb[c];
}

__global__ __launch_bounds__(256) void concat_kernel(
    const float* __restrict__ e, const float* __restrict__ g,
    float* __restrict__ h, long total)
{
  const long i = (long)blockIdx.x * 256 + threadIdx.x;
  if (i >= total) return;
  const long r = i >> 9;
  const int  c = (int)(i & 511);
  h[i] = (c < 256) ? e[r * 256 + c] : g[r * 256 + (c - 256)];
}

__global__ __launch_bounds__(256) void bn_relu_res_kernel(
    const float* __restrict__ conv, const float* __restrict__ g,
    const float* __restrict__ b, float* __restrict__ h, int C, long total)
{
  const long i = (long)blockIdx.x * 256 + threadIdx.x;
  if (i >= total) return;
  const int c = (int)(i % C);
  const float sc = g[c] * rsqrtf(1.0f + 1e-5f);
  h[i] += fmaxf(conv[i] * sc + b[c], 0.0f);
}

__global__ __launch_bounds__(256) void maxpool_kernel(
    const float* __restrict__ h, float* __restrict__ mp, int L)
{
  const int idx = blockIdx.x * 256 + threadIdx.x;
  const int b = idx >> 9;
  const int c = idx & 511;
  float m = -3.4e38f;
  for (int l = 0; l < L; ++l)
    m = fmaxf(m, h[(long)(b * L + l) * 512 + c]);
  mp[idx] = m;
}

__global__ __launch_bounds__(256) void out_kernel(
    const float* __restrict__ mp, const float* __restrict__ w,
    const float* __restrict__ bo, float* __restrict__ out)
{
  const int b = blockIdx.x;
  __shared__ float red[256];
  const int t = threadIdx.x;
  float s = 0.0f;
  for (int c = t; c < 512; c += 256) s += mp[b * 512 + c] * w[c];
  red[t] = s;
  __syncthreads();
  for (int o = 128; o > 0; o >>= 1) {
    if (t < o) red[t] += red[t + o];
    __syncthreads();
  }
  if (t == 0) out[b] = red[0] + bo[0];
}

// ---------------------------------------------------------------------------
// Host orchestration
// ---------------------------------------------------------------------------
static void launch_gemm_dense(hipStream_t st, const float* A, int lda,
                              const float* W, long ldw,
                              float* Out, int ldo, int M, int N, int K)
{
  const int blocks = (M / 128) * (N / 16);
  gemm_bf16_wmma_tdm<<<blocks, 128, 0, st>>>(A, lda, W, ldw, Out, ldo, M, N, K);
}

static void launch_gemm_gen(hipStream_t st, const float* A, int lda,
                            const float* W, long ldw, int wstride,
                            float* Out, int ldo, int M, int N, int K, int L,
                            int shift, int accumulate)
{
  const int blocks = (M / 32) * (N / 16) / 4;
  if ((K & 31) == 0)
    gemm_bf16_wmma_gen<true><<<blocks, 128, 0, st>>>(
        A, lda, W, ldw, wstride, Out, ldo, M, N, K, L, shift, accumulate);
  else
    gemm_bf16_wmma_gen<false><<<blocks, 128, 0, st>>>(
        A, lda, W, ldw, wstride, Out, ldo, M, N, K, L, shift, accumulate);
}

struct MambaBufs { float *xi, *z, *xc, *dtb, *dbl, *tmp; };

static void run_mamba(hipStream_t st, float* h, int D, int di, int dtr,
                      const float* const* p,  // 9 param ptrs (mamba dict order)
                      const float* ln_g, const float* ln_b,
                      const MambaBufs& w, int M, int L, int B)
{
  const int nd = dtr + 2 * 16;
  const long tot = (long)M * di;
  const int blk = (int)((tot + 255) / 256);
  // in_proj: xi = h @ Wi[0:di]^T ; z = h @ Wi[di:2di]^T
  launch_gemm_dense(st, h, D, p[0],                D, w.xi, di, M, di, D);
  launch_gemm_dense(st, h, D, p[0] + (long)di * D, D, w.z,  di, M, di, D);
  // causal depthwise conv + SiLU
  dwconv_silu_kernel<<<blk, 256, 0, st>>>(w.xi, p[1], p[2], w.xc, L, di, tot);
  // x_proj: dbl = xc @ Wx^T  (nd = dtr + 32)
  launch_gemm_dense(st, w.xc, di, p[3], di, w.dbl, nd, M, nd, di);
  // dt_proj on first dtr columns of dbl (K = 16/32), then softplus(+bias)
  launch_gemm_gen(st, w.dbl, nd, p[4], dtr, 1, w.dtb, di, M, di, dtr, L, 0, 0);
  bias_softplus_kernel<<<blk, 256, 0, st>>>(w.dtb, p[5], di, tot);
  // selective scan (writes y into xi buffer, which is free now)
  dim3 sg(di / 256, B);
  scan_kernel<<<sg, 256, 0, st>>>(w.dtb, w.xc, w.dbl, nd, dtr,
                                  p[6], p[7], w.z, w.xi, L, di);
  // out_proj + LN + residual
  launch_gemm_dense(st, w.xi, di, p[8], di, w.tmp, D, M, D, di);
  ln_residual_kernel<<<M, 256, 0, st>>>(w.tmp, ln_g, ln_b, h, D);
}

extern "C" void kernel_launch(void* const* d_in, const int* in_sizes, int n_in,
                              void* d_out, int out_size, void* d_ws, size_t ws_size,
                              hipStream_t stream)
{
  (void)in_sizes; (void)n_in; (void)out_size; (void)ws_size;
  const int B = 4, L = 1024, M = B * L;

  const float* x = (const float*)d_in[0];
  auto in = [&](int i) { return (const float*)d_in[i]; };

  // Workspace layout (floats), sized for the largest Mamba (di = 2048).
  float* w0 = (float*)d_ws;
  float* e    = w0;                  w0 += (long)M * 256;
  float* g    = w0;                  w0 += (long)M * 256;
  float* h    = w0;                  w0 += (long)M * 512;
  float* tmp  = w0;                  w0 += (long)M * 512;
  float* xi   = w0;                  w0 += (long)M * 2048;   // also scan output y
  float* z    = w0;                  w0 += (long)M * 2048;
  float* xc   = w0;                  w0 += (long)M * 2048;
  float* dtb  = w0;                  w0 += (long)M * 2048;
  float* dbl  = w0;                  w0 += (long)M * 96;
  float* mp   = w0;                  w0 += B * 512;

  MambaBufs mb{xi, z, xc, dtb, dbl, tmp};

  // ---- front end: modality projection + MMCNN encoders -> e, g (M x 256)
  frontend_kernel<<<M, 256, 0, stream>>>(
      x, in(1), in(2), in(3), in(4), in(5), in(6), in(7), in(8),
      in(9), in(10), e, g, L);

  // ---- per-modality Mamba blocks (D=256, di=1024, dtr=16)
  {
    const float* pe[9] = { in(11), in(12), in(13), in(14), in(15),
                           in(16), in(17), in(18), in(19) };
    run_mamba(stream, e, 256, 1024, 16, pe, in(29), in(30), mb, M, L, B);
    const float* pg[9] = { in(20), in(21), in(22), in(23), in(24),
                           in(25), in(26), in(27), in(28) };
    run_mamba(stream, g, 256, 1024, 16, pg, in(31), in(32), mb, M, L, B);
  }

  // ---- concat -> h (M x 512)
  {
    const long tot = (long)M * 512;
    concat_kernel<<<(int)((tot + 255) / 256), 256, 0, stream>>>(e, g, h, tot);
  }

  // ---- enssm conv1d(512->512,k=3,same) as 3 shifted accumulating GEMMs
  {
    const float* cw = in(33);   // (512, 512, 3): tap s at cw + s, wstride 3
    launch_gemm_gen(stream, h, 512, cw + 0, 512L * 3, 3, tmp, 512, M, 512, 512, L, -1, 0);
    launch_gemm_gen(stream, h, 512, cw + 1, 512L * 3, 3, tmp, 512, M, 512, 512, L,  0, 1);
    launch_gemm_gen(stream, h, 512, cw + 2, 512L * 3, 3, tmp, 512, M, 512, 512, L, +1, 1);
    const long tot = (long)M * 512;
    bn_relu_res_kernel<<<(int)((tot + 255) / 256), 256, 0, stream>>>(
        tmp, in(34), in(35), h, 512, tot);
  }

  // ---- fused Mamba block (D=512, di=2048, dtr=32)
  {
    const float* pm[9] = { in(36), in(37), in(38), in(39), in(40),
                           in(41), in(42), in(43), in(44) };
    run_mamba(stream, h, 512, 2048, 32, pm, in(45), in(46), mb, M, L, B);
  }

  // ---- global max pool over L + final linear
  maxpool_kernel<<<(B * 512) / 256, 256, 0, stream>>>(h, mp, L);
  out_kernel<<<B, 256, 0, stream>>>(mp, in(47), in(48), (float*)d_out);
}
